// Conv3d_76141180223681
// MI455X (gfx1250) — matrix operand order repaired and checked by reading, not compiled, not run on hardware
//
#include <hip/hip_runtime.h>

typedef __attribute__((ext_vector_type(16))) _Float16 v16h;
typedef __attribute__((ext_vector_type(8)))  _Float16 v8h;
typedef __attribute__((ext_vector_type(8)))  float    v8f;

// Workspace layout (bytes):
//  xh : f16 x, channel-last [n][d*h*w][ic]          16,777,216 B
//  wh : f16 w, [tap(27)][oc(128)][ic(256)]           1,769,472 B
//  yh : f16 y, [n][oc][33][33][33]                  73,598,976 B  (fits in 192MB L2)
#define XH_OFF 0
#define WH_OFF 16777216
#define YH_OFF 18546688

// ---------------- Prep 1: x [n][ic][4096] f32 -> xh [n][sp][ic] f16 (LDS-tiled transpose)
__global__ void k_prep_x(const float* __restrict__ x, _Float16* __restrict__ xh) {
    __shared__ float t[32][33];
    int b = blockIdx.x;              // 8 n * 8 icT * 128 spT = 8192
    int spT = b % 128; b /= 128;
    int icT = b % 8;   b /= 8;
    int n   = b;
    int tx = threadIdx.x & 31, ty = threadIdx.x >> 5;   // 32 x 8
    int ic0 = icT * 32, sp0 = spT * 32;
    const float* src = x + ((size_t)n * 256 + ic0) * 4096 + sp0;
#pragma unroll
    for (int j = 0; j < 4; ++j) {
        int r = ty * 4 + j;                      // ic row, coalesced over sp
        t[r][tx] = src[(size_t)r * 4096 + tx];
    }
    __syncthreads();
    _Float16* dst = xh + ((size_t)n * 4096 + sp0) * 256 + ic0;
#pragma unroll
    for (int j = 0; j < 4; ++j) {
        int s = ty * 4 + j;                      // spatial row, coalesced over ic
        dst[(size_t)s * 256 + tx] = (_Float16)t[tx][s];
    }
}

// ---------------- Prep 2: w [oc][ic][27] f32 -> wh [tap][oc][ic] f16
__global__ void k_prep_w(const float* __restrict__ w, _Float16* __restrict__ wh) {
    int i = blockIdx.x * 256 + threadIdx.x;      // 27*128*256 = 884736
    int ic  = i & 255;
    int oc  = (i >> 8) & 127;
    int tap = i >> 15;                            // 0..26
    wh[i] = (_Float16)w[((size_t)oc * 256 + ic) * 27 + tap];
}

// ---------------- Main: transposed conv as parity-split implicit GEMM on WMMA.
// Block = 16 same-parity output voxels (fixed n,od,oh) x 128 oc (8 waves x 16).
// The block's x working set (<= 2x2x17 rows of 256 f16 ch) is DMA'd once into
// LDS with global_load_async_to_lds_b128; halo rows outside the volume are
// zero-filled with ds stores in the same pass, so B reads are unconditional.
__global__ void k_tconv_wmma(const _Float16* __restrict__ xh,
                             const _Float16* __restrict__ wh,
                             _Float16* __restrict__ yh) {
    __shared__ __align__(16) _Float16 xs[2][2][17][256];   // 34,816 B

    int b = blockIdx.x;                 // 8 n * 33 od * 33 oh * 3 xtile = 26136
    int xt = b % 3;  b /= 3;
    int oh = b % 33; b /= 33;
    int od = b % 33;
    int n  = b / 33;

    int lane = threadIdx.x & 31;
    int wv   = threadIdx.x >> 5;        // 8 waves -> 8 oc tiles of 16
    int oc0  = wv * 16;
    int col  = lane & 15;               // N column / A's M row
    int hi   = lane >> 4;               // lane-half select

    int rz = od & 1, zi = od >> 1;
    int ry = oh & 1, yi = oh >> 1;
    int rx  = (xt == 2) ? 1 : 0;        // xtile 0,1: even ow; xtile 2: odd ow
    int xi0 = (xt == 1) ? 16 : 0;
    int xi  = xi0 + col;

    int nzv = rz ? 1 : 2;
    int nyv = ry ? 1 : 2;
    int nx  = rx ? 1 : 2;
    int iwBase = rx ? 0 : (xi0 - 1);

    // ---- Async DMA of the x tile into LDS; invalid halo rows -> zero ds stores.
    int nrows = nzv * nyv * 17;
    for (int t = threadIdx.x; t < nrows * 32; t += 256) {
        int rowi  = t >> 5;
        int chunk = t & 31;                               // 16B chunk in 512B row
        int wl  = rowi % 17;
        int rzy = rowi / 17;
        int yv  = rzy % nyv;
        int zv  = rzy / nyv;
        int id  = rz ? zi : (zi - 1 + zv);
        int ih  = ry ? yi : (yi - 1 + yv);
        int iw  = iwBase + wl;
        _Float16* ldsp = &xs[zv][yv][wl][chunk * 8];
        if (((unsigned)id < 16u) & ((unsigned)ih < 16u) & ((unsigned)iw < 16u)) {
            unsigned lds = (unsigned)(uintptr_t)ldsp;
            unsigned long long g = (unsigned long long)(xh +
                (((size_t)n * 4096 + id * 256 + ih * 16 + iw) * 256 + chunk * 8));
            asm volatile("global_load_async_to_lds_b128 %0, %1, off"
                         :: "v"(lds), "v"(g) : "memory");
        } else {
            v8h z = {};
            *(v8h*)ldsp = z;                              // zero halo row chunk
        }
    }
    asm volatile("s_wait_asynccnt 0x0" ::: "memory");
    __syncthreads();

    v8f acc = {};

    for (int tz = 0; tz < nzv; ++tz) {
        int id = rz ? zi : (zi - 1 + tz);
        int kd = rz ? 1  : 2 * tz;
        if (id < 0 || id > 15) continue;                 // wave-uniform
        for (int tyk = 0; tyk < nyv; ++tyk) {
            int ih = ry ? yi : (yi - 1 + tyk);
            int kh = ry ? 1  : 2 * tyk;
            if (ih < 0 || ih > 15) continue;             // wave-uniform
            for (int txk = 0; txk < nx; ++txk) {
                int kw = rx ? 1 : 2 * txk;
                int wl = rx ? col : (col + txk);         // LDS row, always in range
                int tap = (kd * 3 + kh) * 3 + kw;
                // B from LDS: lanes<16 K {0..7,16..23}, >=16 {8..15,24..31}
                const _Float16* pB = &xs[tz][tyk][wl][hi * 8];
                // A from global (L2-resident): lanes<16 K {0..7,16..23}, >=16 {8..15,24..31}
                const _Float16* pA = wh + ((size_t)tap * 128 + oc0 + col) * 256 + hi * 8;
#pragma unroll
                for (int icc = 0; icc < 256; icc += 32) {
                    v8h a0 = *(const v8h*)(pA + icc);          // K = icc+e+8*hi
                    v8h a1 = *(const v8h*)(pA + icc + 16);     // K = icc+e+16+8*hi
                    v8h b0 = *(const v8h*)(pB + icc);          // K = icc+e+8*hi
                    v8h b1 = *(const v8h*)(pB + icc + 16);
                    v16h a  = __builtin_shufflevector(a0, a1, 0,1,2,3,4,5,6,7,
                                                              8,9,10,11,12,13,14,15);
                    v16h bf = __builtin_shufflevector(b0, b1, 0,1,2,3,4,5,6,7,
                                                              8,9,10,11,12,13,14,15);
                    acc = __builtin_amdgcn_wmma_f32_16x16x32_f16(
                        false, a, false, bf, (short)0, acc, false, false);
                }
            }
        }
    }

    // C/D layout: element (M = v + 8*hi, N = col)
    int ow = 2 * xi + rx;
    if (ow < 33) {
        size_t base = (((size_t)n * 128 + oc0 + hi * 8) * 35937) + (size_t)od * 1089 + oh * 33 + ow;
#pragma unroll
        for (int v = 0; v < 8; ++v)
            yh[base + (size_t)v * 35937] = (_Float16)acc[v];
    }
}

// ---------------- FIR (4x4x4 separable [1,3,3,1]/64) + bias; reads L2-resident yh
__global__ void k_fir(const _Float16* __restrict__ yh,
                      const float* __restrict__ bias,
                      float* __restrict__ out) {
    __shared__ float t[7][11][35];          // z:7 y:11 x:35 input halo tile
    int b = blockIdx.x;                     // 8 n * 128 c * 8 zt * 4 yt = 32768
    int yt = b & 3;   b >>= 2;
    int zt = b & 7;   b >>= 3;
    int c  = b & 127; b >>= 7;
    int n  = b;
    int z0 = zt * 4, y0 = yt * 8;

    const _Float16* src = yh + ((size_t)n * 128 + c) * 35937;
    for (int idx = threadIdx.x; idx < 7 * 11 * 35; idx += 256) {
        int zz = idx / 385;
        int r  = idx - zz * 385;
        int yy = r / 35;
        int xx = r - yy * 35;
        int gz = z0 - 1 + zz, gy = y0 - 1 + yy, gx = xx - 1;
        float v = 0.f;
        if ((unsigned)gz < 33u && (unsigned)gy < 33u && (unsigned)gx < 33u)
            v = (float)src[(size_t)gz * 1089 + gy * 33 + gx];
        ((float*)t)[idx] = v;
    }
    __syncthreads();

    int tx = threadIdx.x & 31;              // output x
    int ty = threadIdx.x >> 5;              // output y (0..7)
    const float K1[4] = {1.f, 3.f, 3.f, 1.f};
    float cs[7];
#pragma unroll
    for (int zz = 0; zz < 7; ++zz) {        // y,x-filtered column sums
        float s = 0.f;
#pragma unroll
        for (int bb = 0; bb < 4; ++bb) {
            float r = 0.f;
#pragma unroll
            for (int cc = 0; cc < 4; ++cc)
                r += K1[cc] * t[zz][ty + bb][tx + cc];
            s += K1[bb] * r;
        }
        cs[zz] = s;
    }
    float bv = bias[c];
    size_t ob = ((size_t)n * 128 + c) * 32768 + (size_t)z0 * 1024 + (y0 + ty) * 32 + tx;
#pragma unroll
    for (int za = 0; za < 4; ++za) {
        float s = K1[0]*cs[za] + K1[1]*cs[za+1] + K1[2]*cs[za+2] + K1[3]*cs[za+3];
        out[ob + (size_t)za * 1024] = s * (1.f / 64.f) + bv;
    }
}

extern "C" void kernel_launch(void* const* d_in, const int* in_sizes, int n_in,
                              void* d_out, int out_size, void* d_ws, size_t ws_size,
                              hipStream_t stream) {
    const float* x    = (const float*)d_in[0];   // [8,256,16,16,16]
    const float* w    = (const float*)d_in[1];   // [128,256,3,3,3]
    const float* bias = (const float*)d_in[2];   // [128]
    float* out = (float*)d_out;                  // [8,128,32,32,32]
    char* ws = (char*)d_ws;
    _Float16* xh = (_Float16*)(ws + XH_OFF);
    _Float16* wh = (_Float16*)(ws + WH_OFF);
    _Float16* yh = (_Float16*)(ws + YH_OFF);

    hipLaunchKernelGGL(k_prep_x,     dim3(8192),        dim3(256), 0, stream, x, xh);
    hipLaunchKernelGGL(k_prep_w,     dim3(3456),        dim3(256), 0, stream, w, wh);
    hipLaunchKernelGGL(k_tconv_wmma, dim3(8*33*33*3),   dim3(256), 0, stream, xh, wh, yh);
    hipLaunchKernelGGL(k_fir,        dim3(8*128*8*4),   dim3(256), 0, stream, yh, bias, out);
}
